// QuantizedLinear_60584808677418
// MI455X (gfx1250) — compile-verified, matrix operand order checked by reading, not run on hardware
//
#include <hip/hip_runtime.h>
#include <hip/hip_bf16.h>

typedef __attribute__((ext_vector_type(16))) _Float16 v16h;
typedef __attribute__((ext_vector_type(8)))  _Float16 v8h;
typedef __attribute__((ext_vector_type(2)))  _Float16 h2;
typedef __attribute__((ext_vector_type(8)))  float    v8f;

namespace {
constexpr int KDIM   = 4096;
constexpr int NDIM   = 14336;
constexpr int MDIM   = 256;
constexpr int QGROUP = 128;              // values per quant group
constexpr int KCH    = 32;               // K per WMMA
constexpr int NGRP   = KDIM / QGROUP;    // 32 groups
constexpr int NCHUNK = KDIM / KCH;       // 128 k-chunks
constexpr int LDSW   = 40;               // padded halves per A row (32 + 8) -> 80B stride
constexpr int MTILE  = 128;              // rows per workgroup
constexpr int NTILE  = 128;              // cols per workgroup (4 waves x 32)
constexpr int NSUB   = 8;                // M sub-tiles per wave
}

union AFrag {
  v16h v;
  struct { v8h lo, hi; } p;
};

union BFrag {
  v16h v;
  h2   p[8];
};

union H2U {
  unsigned int u;
  h2 h;
};

// Dequant 2 int4 values (one packed byte, selected by j from dword q) into a
// packed f16 pair: v_perm_b32 + v_and_or + v_pk_fma_f16.
//   t = byte j of q replicated into result bytes 0 and 2
//   d = (t & 0x00F0000F) | 0x64006400   -> f16 pair (1024+lo, 1024+16*hi)
//   w = d * (s, s/16) + (-1032*s, -72*s) = ((lo-8)*s, (hi-8)*s)
__device__ __forceinline__ h2 dq_pair(unsigned int q, int j, h2 scl, h2 bia) {
  const unsigned int sel = 0x0C000C00u + (unsigned int)j * 0x00010001u;
  const unsigned int t = __builtin_amdgcn_perm(0u, q, sel);
  H2U d;
  d.u = (t & 0x00F0000Fu) | 0x64006400u;
  return d.h * scl + bia;   // v_pk_fma_f16
}

// Build one 16x32 f16 B fragment (this lane's K-slice) from two packed dwords.
__device__ __forceinline__ v16h build_b(unsigned int q0, unsigned int q1,
                                        h2 scl, h2 bia) {
  BFrag bf;
#pragma unroll
  for (int j = 0; j < 4; ++j) {
    bf.p[j]     = dq_pair(q0, j, scl, bia);
    bf.p[4 + j] = dq_pair(q1, j, scl, bia);
  }
  return bf.v;
}

// Convert 8 consecutive fp32 x-values to f16, store to LDS (1x ds_store_b128).
__device__ __forceinline__ void fill_a8(_Float16* dst, const float* src) {
  const float4* s4 = (const float4*)src;
  float4 f0 = s4[0], f1 = s4[1];
  v8h h;
  h[0] = (_Float16)f0.x; h[1] = (_Float16)f0.y; h[2] = (_Float16)f0.z; h[3] = (_Float16)f0.w;
  h[4] = (_Float16)f1.x; h[5] = (_Float16)f1.y; h[6] = (_Float16)f1.z; h[7] = (_Float16)f1.w;
  *(v8h*)dst = h;
}

__global__ __launch_bounds__(128)
void qlinear_wmma_kernel(const float* __restrict__ x,
                         const unsigned char* __restrict__ wq,
                         const float* __restrict__ scales,
                         const float* __restrict__ bias,
                         float* __restrict__ out)
{
  // Double-buffered A tile: 128 rows x 32 K (f16), padded stride (80B) for
  // bank-conflict-free ds_load_b128 fragment reads.  2 * 128*40*2 = 20 KB.
  __shared__ _Float16 As[2][MTILE * LDSW];

  const int tid  = threadIdx.x;
  const int wave = tid >> 5;       // 0..3 : owns 32 output columns
  const int lane = tid & 31;
  const int l16  = lane & 15;
  const int kh   = lane >> 4;      // K-half selector per WMMA 16-bit layout

  const int nBase = blockIdx.x * NTILE;
  const int mBase = blockIdx.y * MTILE;

  // Two output columns owned by this lane (B-matrix: lane -> N, K along VGPRs).
  const int o0 = nBase + wave * 32 + l16;
  const int o1 = o0 + 16;
  const unsigned char* wrow0 = wq + (size_t)o0 * (KDIM / 2);
  const unsigned char* wrow1 = wq + (size_t)o1 * (KDIM / 2);
  const float* srow0 = scales + (size_t)o0 * NGRP;
  const float* srow1 = scales + (size_t)o1 * NGRP;

  // Cooperative A staging: thread t covers rows (t>>2)+{0,32,64,96},
  // K-quarter t&3 (8 fp32 -> 8 f16 each pass); 128B-contiguous global reads.
  const int arow = tid >> 2;
  const int aq   = tid & 3;
  const float* xrow = x + (size_t)(mBase + arow) * KDIM + aq * 8;
  const int aoff = arow * LDSW + aq * 8;

  v8f acc0[NSUB] = {};   // column o0
  v8f acc1[NSUB] = {};   // column o1

#pragma unroll
  for (int r = 0; r < 4; ++r)      // prime k-chunk 0
    fill_a8(&As[0][aoff + r * 32 * LDSW], xrow + (size_t)r * 32 * KDIM);
  __syncthreads();

  for (int g = 0; g < NGRP; ++g) {
    const float s0 = srow0[g];
    const float s1 = srow1[g];
    H2U scl0, bia0, scl1, bia1;
    scl0.h[0] = (_Float16)s0;              scl0.h[1] = (_Float16)(s0 * 0.0625f);
    bia0.h[0] = (_Float16)(-1032.0f * s0); bia0.h[1] = (_Float16)(-72.0f * s0);
    scl1.h[0] = (_Float16)s1;              scl1.h[1] = (_Float16)(s1 * 0.0625f);
    bia1.h[0] = (_Float16)(-1032.0f * s1); bia1.h[1] = (_Float16)(-72.0f * s1);

    if (g + 1 < NGRP) {
      // global_prefetch_b8 for next group's packed int4 bytes
      __builtin_prefetch(wrow0 + (((g + 1) * QGROUP) >> 1), 0, 1);
      __builtin_prefetch(wrow1 + (((g + 1) * QGROUP) >> 1), 0, 1);
    }

#pragma unroll
    for (int c = 0; c < QGROUP / KCH; ++c) {
      const int kci = g * (QGROUP / KCH) + c;
      const int buf = kci & 1;

      // ISA 16-bit B layout (mirror of A): lanes 0-15 hold K kc+0..7 / kc+16..23,
      // lanes 16-31 hold K kc+8..15 / kc+24..31, packed 2 per VGPR.
      const unsigned int* pk0 = (const unsigned int*)(wrow0 + kci * 16);
      const unsigned int* pk1 = (const unsigned int*)(wrow1 + kci * 16);
      const unsigned int q00 = pk0[kh];
      const unsigned int q01 = pk0[kh + 2];
      const unsigned int q10 = pk1[kh];
      const unsigned int q11 = pk1[kh + 2];

      // Stage next A chunk into the other LDS buffer (overlaps with dequant).
      if (kci + 1 < NCHUNK) {
        const float* xn = xrow + (kci + 1) * KCH;
        _Float16* an = &As[buf ^ 1][aoff];
#pragma unroll
        for (int r = 0; r < 4; ++r)
          fill_a8(an + r * 32 * LDSW, xn + (size_t)r * 32 * KDIM);
      }

      const v16h bf0 = build_b(q00, q01, scl0.h, bia0.h);
      const v16h bf1 = build_b(q10, q11, scl1.h, bia1.h);

      // 8 M-subtiles; each A fragment feeds 2 WMMAs (both B fragments) and each
      // B fragment feeds 8 WMMAs.  Pipeline A ds_loads one sub-tile ahead.
      const _Float16* abuf = &As[buf][(size_t)l16 * LDSW + kh * 8];
      v8h alo = *(const v8h*)abuf;
      v8h ahi = *(const v8h*)(abuf + 16);
#pragma unroll
      for (int sub = 0; sub < NSUB; ++sub) {
        AFrag af;
        af.p.lo = alo;
        af.p.hi = ahi;
        if (sub < NSUB - 1) {
          const _Float16* ap = abuf + (sub + 1) * 16 * LDSW;
          alo = *(const v8h*)ap;
          ahi = *(const v8h*)(ap + 16);
        }
        acc0[sub] = __builtin_amdgcn_wmma_f32_16x16x32_f16(
            false, af.v, false, bf0, (short)0, acc0[sub], false, false);
        acc1[sub] = __builtin_amdgcn_wmma_f32_16x16x32_f16(
            false, af.v, false, bf1, (short)0, acc1[sub], false, false);
      }
      __syncthreads();  // next-buffer fill visible; current buffer safe to overwrite
    }
  }

  // Epilogue: C/D layout -> VGPR r holds M = r + 8*kh, column = o0 / o1.
  const float bi0 = bias[o0];
  const float bi1 = bias[o1];
#pragma unroll
  for (int sub = 0; sub < NSUB; ++sub) {
    const int r0 = mBase + sub * 16 + kh * 8;
#pragma unroll
    for (int r = 0; r < 8; ++r) {
      out[(size_t)(r0 + r) * NDIM + o0] = acc0[sub][r] + bi0;
      out[(size_t)(r0 + r) * NDIM + o1] = acc1[sub][r] + bi1;
    }
  }
}

extern "C" void kernel_launch(void* const* d_in, const int* in_sizes, int n_in,
                              void* d_out, int out_size, void* d_ws, size_t ws_size,
                              hipStream_t stream) {
  const float* x          = (const float*)d_in[0];
  const unsigned char* wq = (const unsigned char*)d_in[1];
  const float* scales     = (const float*)d_in[2];
  const float* bias       = (const float*)d_in[3];
  float* out              = (float*)d_out;

  dim3 grid(NDIM / NTILE, MDIM / MTILE);   // (112, 2)
  qlinear_wmma_kernel<<<grid, dim3(128), 0, stream>>>(x, wq, scales, bias, out);
}
